// RNN_58480274702936
// MI455X (gfx1250) — matrix operand order repaired and checked by reading, not compiled, not run on hardware
//
#include <hip/hip_runtime.h>
#include <hip/hip_bf16.h>
#include <math.h>

typedef __attribute__((ext_vector_type(16))) __bf16 v16bf;
typedef __attribute__((ext_vector_type(8)))  __bf16 v8bf;
typedef __attribute__((ext_vector_type(8)))  float  v8f;
typedef __attribute__((ext_vector_type(4)))  unsigned u32x4;
typedef __attribute__((ext_vector_type(8)))  int      i32x8;
typedef __attribute__((ext_vector_type(4)))  int      i32x4;

#define BSZ 64
#define TSZ 512
#define DSZ 1024
#define HSZ 1024

#define WH_STRIDE 1032                    // halves; +4 banks/row -> conflict-free
#define WH_BYTES  (32 * WH_STRIDE * 2)    // 66048
#define H_ROW_B   2080                    // 2048B + 2x16B TDM pad per row
#define HT_BYTES  (64 * H_ROW_B)          // 133120

union V16 { v16bf v; v8bf h[2]; };

// ---------------- Phase 0: init h buffers, bf16 Wh copy, barrier ----------
__global__ void rnn_init_kernel(const float* __restrict__ Wh, __bf16* __restrict__ whbf,
                                __bf16* __restrict__ hbufs, int* __restrict__ bar) {
    int i = blockIdx.x * blockDim.x + threadIdx.x;
    if (i < 2 * BSZ * HSZ) hbufs[i] = (__bf16)0.0f;
    if (i < HSZ * DSZ) whbf[i] = (__bf16)Wh[i];
    if (i == 0) { bar[0] = 0; bar[1] = 0; }
}

// ---------------- Phase 1: xi = x * Wi^T + bi (bf16 WMMA, f32 accum) -----
// Block tile M=64 x N=128, 8 waves (4 M-strips x 2 N-halves), 16x64 per wave.
// Ping-pong LDS double buffer: next K-chunk's global loads issue before the
// current chunk's WMMAs so load latency hides under the matrix math.
__global__ void __launch_bounds__(256) rnn_xproj_kernel(
    const float* __restrict__ x, const float* __restrict__ Wi,
    const float* __restrict__ bi, float* __restrict__ xi) {
    __shared__ __bf16 aLds[2][64 * 40];    // stride 40 halves (80B, 16B aligned)
    __shared__ __bf16 bLds[2][128 * 40];

    const int tid  = threadIdx.x;
    const int lane = tid & 31;
    const int wave = tid >> 5;
    const int m0 = blockIdx.y * 64;
    const int n0 = blockIdx.x * 128;
    const int mt = wave & 3;
    const int nh = wave >> 2;
    const int hi16 = lane >> 4;
    const int l15  = lane & 15;

    const int ar[2] = { tid >> 3, (tid + 256) >> 3 };
    const int ac    = (tid & 7) * 4;
    int br[4];
    #pragma unroll
    for (int u = 0; u < 4; ++u) br[u] = (tid + u * 256) >> 3;

    v8f acc[4] = {};
    float4 ra[2], rb[4];

    #pragma unroll
    for (int u = 0; u < 2; ++u)
        ra[u] = *(const float4*)&x[(size_t)(m0 + ar[u]) * DSZ + ac];
    #pragma unroll
    for (int u = 0; u < 4; ++u)
        rb[u] = *(const float4*)&Wi[(size_t)(n0 + br[u]) * DSZ + ac];
    #pragma unroll
    for (int u = 0; u < 2; ++u) {
        __bf16* p = &aLds[0][ar[u] * 40 + ac];
        p[0]=(__bf16)ra[u].x; p[1]=(__bf16)ra[u].y; p[2]=(__bf16)ra[u].z; p[3]=(__bf16)ra[u].w;
    }
    #pragma unroll
    for (int u = 0; u < 4; ++u) {
        __bf16* p = &bLds[0][br[u] * 40 + ac];
        p[0]=(__bf16)rb[u].x; p[1]=(__bf16)rb[u].y; p[2]=(__bf16)rb[u].z; p[3]=(__bf16)rb[u].w;
    }
    __syncthreads();

    for (int kc = 0; kc < 32; ++kc) {
        const int cur = kc & 1;
        const int k1 = (kc + 1) * 32;

        if (kc + 1 < 32) {
            #pragma unroll
            for (int u = 0; u < 2; ++u)
                ra[u] = *(const float4*)&x[(size_t)(m0 + ar[u]) * DSZ + k1 + ac];
            #pragma unroll
            for (int u = 0; u < 4; ++u)
                rb[u] = *(const float4*)&Wi[(size_t)(n0 + br[u]) * DSZ + k1 + ac];
        }

        V16 a;
        const int arow = mt * 16 + l15;
        a.h[0] = *(const v8bf*)&aLds[cur][arow * 40 + hi16 * 8];
        a.h[1] = *(const v8bf*)&aLds[cur][arow * 40 + 16 + hi16 * 8];
        #pragma unroll
        for (int j = 0; j < 4; ++j) {
            V16 b;
            const int brow = nh * 64 + j * 16 + l15;
            b.h[0] = *(const v8bf*)&bLds[cur][brow * 40 + hi16 * 8];
            b.h[1] = *(const v8bf*)&bLds[cur][brow * 40 + 16 + hi16 * 8];
            acc[j] = __builtin_amdgcn_wmma_f32_16x16x32_bf16(
                false, a.v, false, b.v, (short)0, acc[j], false, false);
        }

        if (kc + 1 < 32) {
            #pragma unroll
            for (int u = 0; u < 2; ++u) {
                __bf16* p = &aLds[cur ^ 1][ar[u] * 40 + ac];
                p[0]=(__bf16)ra[u].x; p[1]=(__bf16)ra[u].y; p[2]=(__bf16)ra[u].z; p[3]=(__bf16)ra[u].w;
            }
            #pragma unroll
            for (int u = 0; u < 4; ++u) {
                __bf16* p = &bLds[cur ^ 1][br[u] * 40 + ac];
                p[0]=(__bf16)rb[u].x; p[1]=(__bf16)rb[u].y; p[2]=(__bf16)rb[u].z; p[3]=(__bf16)rb[u].w;
            }
        }
        __syncthreads();
    }

    #pragma unroll
    for (int j = 0; j < 4; ++j) {
        const int col = n0 + nh * 64 + j * 16 + l15;
        const float bias = bi[col];
        #pragma unroll
        for (int r = 0; r < 8; ++r) {
            const int row = m0 + mt * 16 + r + hi16 * 8;
            xi[(size_t)row * HSZ + col] = acc[j][r] + bias;
        }
    }
}

// ---- TDM: load an 8-row slice of h (8 x 1024 bf16) into padded LDS ------
// D# per CDNA5 ISA: group0 = {count, lds_addr, global_addr, type=2};
// group1 = {data_size=2B, pad_enable, pad_interval=256dw, pad_amount=4dw,
//           tensor_dim0=1024, tensor_dim1=8, tile_dim0=1024, tile_dim1=8,
//           dim0_stride=1024}. Groups 2/3 zero (2-D tile).
__device__ __forceinline__ void tdm_load_h_slice(unsigned ldsAddr, const void* gp) {
    const unsigned long long ga = (unsigned long long)(size_t)gp;
    u32x4 g0;
    g0[0] = 1u;                                           // count=1, user mode
    g0[1] = ldsAddr;                                      // LDS byte address
    g0[2] = (unsigned)(ga & 0xFFFFFFFFu);                 // global_addr[31:0]
    g0[3] = (unsigned)((ga >> 32) & 0x01FFFFFFu) | (2u << 30);  // addr[56:32]|type=2
    i32x8 g1;
    g1[0] = (1 << 16) | (1 << 20) | (7 << 22) | (3 << 25);  // 2B elts, pad 4dw/256dw
    g1[1] = (int)(1024u << 16);                             // tensor_dim0 lo16 @63:48
    g1[2] = (int)(8u << 16);                                // tensor_dim1 lo16 @111:80
    g1[3] = (int)(1024u << 16);                             // tile_dim0 @127:112
    g1[4] = 8;                                              // tile_dim1 @143:128
    g1[5] = 1024;                                           // dim0_stride lo32 @207:160
    g1[6] = 0; g1[7] = 0;
    i32x4 z4 = {};
#if defined(__clang_major__) && (__clang_major__ >= 23)
    i32x8 z8 = {};
    __builtin_amdgcn_tensor_load_to_lds(g0, g1, z4, z4, z8, 0);
#else
    __builtin_amdgcn_tensor_load_to_lds(g0, g1, z4, z4, 0);
#endif
}

// ---------------- Phase 2: persistent recurrent scan ----------------
// 32 blocks x 256 threads; block owns 32 output columns. Wh slice async-loaded
// once into LDS (ASYNCcnt); each step h_{t-1} (128 KB) is DMA'd into LDS by
// 8 TDM descriptors (TENSORcnt), then the K-loop is pure ds_load + v_wmma.
__global__ void __launch_bounds__(256) rnn_recurrent_kernel(
    const __bf16* __restrict__ whbf, const float* __restrict__ bh,
    float* __restrict__ out /* xi in, h_t out */,
    __bf16* __restrict__ hbuf0, __bf16* __restrict__ hbuf1,
    int* __restrict__ bar, int nblk) {
    extern __shared__ char smem[];
    __bf16* whLds = (__bf16*)smem;                    // 32 x WH_STRIDE halves
    char*   hLdsB = smem + WH_BYTES;                  // 64 rows x 2080 B (TDM pad)
    const unsigned whLdsOff = (unsigned)(size_t)(void*)smem;
    const unsigned hLdsOff  = whLdsOff + WH_BYTES;

    const int tid  = threadIdx.x;
    const int lane = tid & 31;
    const int wave = tid >> 5;
    const int hi16 = lane >> 4;
    const int l15  = lane & 15;
    const int colbase = blockIdx.x * 32;
    const int mt = wave & 3;      // batch-row tile (0..3)
    const int nt = wave >> 2;     // column tile within block (0..1)

    // one-time: async-copy this block's 32 bf16 Wh rows into padded LDS
    for (int i = tid; i < 4096; i += 256) {           // 32 rows x 128 16B-chunks
        const int r = i >> 7, c8 = (i & 127) * 8;
        const unsigned lds = whLdsOff + (unsigned)(r * WH_STRIDE + c8) * 2;
        const void* g = (const void*)(whbf + (size_t)(colbase + r) * HSZ + c8);
        asm volatile("global_load_async_to_lds_b128 %0, %1, off"
                     :: "v"(lds), "v"(g) : "memory");
    }
    asm volatile("s_wait_asynccnt 0x0" ::: "memory");
    __syncthreads();

    const int arow    = mt * 16 + l15;            // batch row 0..63
    const int bRowLds = nt * 16 + l15;            // LDS row = output column
    const int col     = colbase + nt * 16 + l15;
    const float bias  = bh[col];
    const size_t obase = (size_t)col + (size_t)(mt * 16 + hi16 * 8) * TSZ * HSZ;

    for (int t = 0; t < TSZ; ++t) {
        const __bf16* __restrict__ hprev = (t & 1) ? hbuf1 : hbuf0;
        __bf16* __restrict__ hnext       = (t & 1) ? hbuf0 : hbuf1;

        // each wave issues one TDM descriptor for its 8-row slice of h_{t-1}
        tdm_load_h_slice(hLdsOff + (unsigned)(wave * 8) * H_ROW_B,
                         (const void*)(hprev + (size_t)(wave * 8) * HSZ));

        // issue this step's xi loads now; latency hides under the WMMAs
        float xiv[8];
        #pragma unroll
        for (int r = 0; r < 8; ++r)
            xiv[r] = out[obase + (size_t)r * TSZ * HSZ + (size_t)t * HSZ];

        __builtin_amdgcn_s_wait_tensorcnt(0);
        __syncthreads();

        v8f acc = {};
        #pragma unroll 4
        for (int k0 = 0; k0 < HSZ; k0 += 32) {
            V16 a, b;
            // padded row: +16B gap after first 1024 data bytes (k>=512)
            const char* ap = hLdsB + arow * H_ROW_B + ((k0 + hi16 * 8) << 1)
                                   + ((k0 >> 9) << 4);
            a.h[0] = *(const v8bf*)ap;
            a.h[1] = *(const v8bf*)(ap + 32);
            const __bf16* bp = &whLds[bRowLds * WH_STRIDE + k0 + hi16 * 8];
            b.h[0] = *(const v8bf*)bp;
            b.h[1] = *(const v8bf*)(bp + 16);
            acc = __builtin_amdgcn_wmma_f32_16x16x32_bf16(
                false, a.v, false, b.v, (short)0, acc, false, false);
        }

        // h_t = tanh(Wh-part + xi + bh); fp32 -> out, bf16 -> next h buffer
        #pragma unroll
        for (int r = 0; r < 8; ++r) {
            const int row = mt * 16 + r + hi16 * 8;
            const size_t oidx = obase + (size_t)r * TSZ * HSZ + (size_t)t * HSZ;
            const float v = tanhf(acc[r] + xiv[r] + bias);
            out[oidx] = v;
            hnext[(size_t)row * HSZ + col] = (__bf16)v;
            if (t + 1 < TSZ) __builtin_prefetch(&out[oidx + HSZ], 0, 1);
        }

        // grid-wide sense-reversing barrier (all blocks must see full h_t)
        __threadfence();
        __syncthreads();
        if (tid == 0) {
            int gen = __hip_atomic_load(&bar[1], __ATOMIC_ACQUIRE, __HIP_MEMORY_SCOPE_AGENT);
            int arrived = __hip_atomic_fetch_add(&bar[0], 1, __ATOMIC_ACQ_REL, __HIP_MEMORY_SCOPE_AGENT);
            if (arrived == nblk - 1) {
                __hip_atomic_store(&bar[0], 0, __ATOMIC_RELAXED, __HIP_MEMORY_SCOPE_AGENT);
                __hip_atomic_store(&bar[1], gen + 1, __ATOMIC_RELEASE, __HIP_MEMORY_SCOPE_AGENT);
            } else {
                while (__hip_atomic_load(&bar[1], __ATOMIC_ACQUIRE, __HIP_MEMORY_SCOPE_AGENT) == gen) {
                    __builtin_amdgcn_s_sleep(2);
                }
            }
        }
        __syncthreads();
    }
}

extern "C" void kernel_launch(void* const* d_in, const int* in_sizes, int n_in,
                              void* d_out, int out_size, void* d_ws, size_t ws_size,
                              hipStream_t stream) {
    (void)in_sizes; (void)n_in; (void)out_size; (void)ws_size;
    const float* x  = (const float*)d_in[0];
    const float* Wi = (const float*)d_in[1];
    const float* bi = (const float*)d_in[2];
    const float* Wh = (const float*)d_in[3];
    const float* bh = (const float*)d_in[4];
    float* out = (float*)d_out;

    // workspace: barrier ints | hbuf0 | hbuf1 | bf16 Wh copy
    int*    bar   = (int*)d_ws;
    __bf16* hbuf0 = (__bf16*)((char*)d_ws + 256);
    __bf16* hbuf1 = (__bf16*)((char*)d_ws + 256 + (size_t)BSZ * HSZ * 2);
    __bf16* whbf  = (__bf16*)((char*)d_ws + 512 * 1024);

    // Phase 0: zero h buffers, convert Wh -> bf16, reset barrier
    const int initN = HSZ * DSZ;                          // covers all init work
    rnn_init_kernel<<<(initN + 255) / 256, 256, 0, stream>>>(Wh, whbf, hbuf0, bar);

    // Phase 1: xi = x*Wi^T + bi  -> written into d_out
    dim3 g1(HSZ / 128, (BSZ * TSZ) / 64);
    rnn_xproj_kernel<<<g1, 256, 0, stream>>>(x, Wi, bi, out);

    // Phase 2: persistent recurrent scan (32 blocks, ~195 KB dynamic LDS each)
    const int nblk = HSZ / 32;                            // 32
    const size_t ldsBytes = WH_BYTES + HT_BYTES;          // 199168 B < 320 KB/WGP
    rnn_recurrent_kernel<<<nblk, 256, ldsBytes, stream>>>(whbf, bh, out, hbuf0, hbuf1, bar, nblk);
}
